// AttentiveBPNet_54219667145566
// MI455X (gfx1250) — matrix operand order repaired and checked by reading, not compiled, not run on hardware
//
#include <hip/hip_runtime.h>

typedef __attribute__((ext_vector_type(16))) _Float16 v16h;
typedef __attribute__((ext_vector_type(8)))  float    v8f;
typedef __attribute__((ext_vector_type(4)))  float    v4f;

namespace {
constexpr int N_NODES = 200000;
constexpr int C_DIM   = 64;
constexpr int H_DIM   = 8;
constexpr int M_DIM   = 2;
constexpr int S_DIM   = 16;
constexpr int G_DIM   = 8192;
constexpr float SLOPE = 0.2f;

constexpr int KV_COLS   = 16;              // 8 sk cols + 8 sv cols
constexpr int TILES     = N_NODES / 16;    // 12500 exact
constexpr int WAVES_PB  = 8;               // 256 threads / wave32
}

// ---------------------------------------------------------------------------
// Kernel 1: fold att into W_att -> W2[64 x 16], stored DIRECTLY in the WMMA
// B-operand order this code uses (contiguous k, not the ISA table order; f16), so the GEMM wave loads its 16 halves as
// one contiguous 32-byte v16h:
//   chunk kc (K=32), lane l, elem e:  K = kc*32 + (l>=16)*16 + e, N = l&15
//   Bpk[kc*512 + l*16 + e]
// ---------------------------------------------------------------------------
__global__ void fold_weights_kernel(const float* __restrict__ W_att,
                                    const float* __restrict__ att,
                                    _Float16* __restrict__ Bpk) {
    int o = blockIdx.x * blockDim.x + threadIdx.x;      // 0..1023
    if (o >= C_DIM * KV_COLS) return;
    int k = o >> 4;                // input channel (K dim, 0..63)
    int n = o & 15;                // output column
    int h = n & 7;
    const float* wrow = W_att + (size_t)k * (H_DIM * C_DIM) + h * C_DIM;
    const float* arow = att + h * (2 * C_DIM) + ((n >> 3) << 6);
    float s = 0.f;
#pragma unroll
    for (int cc = 0; cc < C_DIM; ++cc) s = fmaf(wrow[cc], arow[cc], s);

    // scatter into WMMA B register layout
    const int kc   = k >> 5;           // K-chunk (0/1)
    const int kk   = k & 31;
    const int hi   = kk >> 4;          // 0 -> lanes 0-15, 1 -> lanes 16-31
    const int e    = kk & 15;
    const int lane = n + hi * 16;
    Bpk[kc * 512 + lane * 16 + e] = (_Float16)s;
}

// ---------------------------------------------------------------------------
// Kernel 2: skv(N x 16) = x(N x 64) @ W2(64 x 16) via v_wmma_f32_16x16x32_f16.
// One wave -> one 16x16 output tile, K=64 as two K=32 WMMA steps.
// x is streamed once -> non-temporal loads (keep skv resident in L2 instead).
// ---------------------------------------------------------------------------
__global__ void __launch_bounds__(256)
skv_gemm_kernel(const float* __restrict__ x,
                const _Float16* __restrict__ Bpk,
                float* __restrict__ skv) {
    const int wave = threadIdx.x >> 5;                  // 0..7 (wave-uniform)
    const int lane = threadIdx.x & 31;
    const int tile = blockIdx.x * WAVES_PB + wave;
    if (tile >= TILES) return;                          // wave-uniform exit

    const int r0    = tile * 16;
    const int hi    = (lane >= 16) ? 1 : 0;
    const int col   = lane & 15;                        // D column
    const int arow  = lane & 15;                        // A row within tile
    const v4f* xv   = reinterpret_cast<const v4f*>(x + (size_t)(r0 + arow) * C_DIM);

    v8f c = {};                                         // f32 accumulator
#pragma unroll
    for (int kc = 0; kc < 2; ++kc) {                    // K chunks of 32
        // ---- A operand: contiguous k order, same as Bpk ----
        // a[e]   <- x[row][kc*32 + hi*16 + e]       (e = 0..15)
        const int q = kc * 8 + hi * 4;                  // float4 index of abase
        v4f p0 = __builtin_nontemporal_load(xv + q);
        v4f p1 = __builtin_nontemporal_load(xv + q + 1);
        v4f p2 = __builtin_nontemporal_load(xv + q + 2);
        v4f p3 = __builtin_nontemporal_load(xv + q + 3);

        v16h a;
#pragma unroll
        for (int e = 0; e < 4; ++e) {
            a[e]      = (_Float16)p0[e];
            a[4 + e]  = (_Float16)p1[e];
            a[8 + e]  = (_Float16)p2[e];
            a[12 + e] = (_Float16)p3[e];
        }

        // ---- B operand: prepacked in register layout, 32B contiguous ----
        v16h b = *reinterpret_cast<const v16h*>(Bpk + kc * 512 + lane * 16);

        c = __builtin_amdgcn_wmma_f32_16x16x32_f16(
                /*neg_a=*/false, a, /*neg_b=*/false, b,
                /*c_mod=*/(short)0, c, /*reuse_a=*/false, /*reuse_b=*/false);
    }

    // ---- D: VGPR i -> row r0 + i + hi*8, col = lane&15 ----
    const int rbase = r0 + hi * 8;
#pragma unroll
    for (int i = 0; i < 8; ++i)
        skv[(size_t)(rbase + i) * KV_COLS + col] = c[i];
}

// ---------------------------------------------------------------------------
// Kernel 3: gather + leaky_relu + mean over S + softmax over m2.
// Thread t -> (g, m1, h);  h fastest so 8 threads read one skv row coalesced.
// out[g, m1, m2, h] flat = ((g*2 + m1)*2 + m2)*8 + h
// ---------------------------------------------------------------------------
__global__ void __launch_bounds__(256)
score_kernel(const float* __restrict__ skv,
             const int* __restrict__ nidx,     // (G, M, 2, S) int32
             float* __restrict__ out) {
    int t = blockIdx.x * blockDim.x + threadIdx.x;
    if (t >= G_DIM * M_DIM * H_DIM) return;
    const int h  = t & 7;
    const int m1 = (t >> 3) & 1;
    const int g  = t >> 4;

    // sel=1 -> idx_k, sel=0 -> idx_v
    const int* idxk = nidx + (((size_t)g * M_DIM + m1) * 2 + 1) * S_DIM;

    float skvals[S_DIM];
#pragma unroll
    for (int s = 0; s < S_DIM; ++s)
        skvals[s] = skv[(size_t)idxk[s] * KV_COLS + h];

    float avg[M_DIM];
#pragma unroll
    for (int m2 = 0; m2 < M_DIM; ++m2) {
        const int* idxv = nidx + (((size_t)g * M_DIM + m2) * 2 + 0) * S_DIM;
        float acc = 0.f;
#pragma unroll
        for (int s = 0; s < S_DIM; ++s) {
            float v = skvals[s] + skv[(size_t)idxv[s] * KV_COLS + 8 + h];
            acc += (v >= 0.f) ? v : SLOPE * v;          // leaky_relu
        }
        avg[m2] = acc * (1.0f / (float)S_DIM);
    }

    const float mx  = fmaxf(avg[0], avg[1]);
    const float e0  = __expf(avg[0] - mx);
    const float e1  = __expf(avg[1] - mx);
    const float inv = 1.0f / (e0 + e1);

    const size_t ob = (((size_t)g * M_DIM + m1) * M_DIM) * H_DIM + h;
    out[ob]          = e0 * inv;   // m2 = 0
    out[ob + H_DIM]  = e1 * inv;   // m2 = 1
}

// ---------------------------------------------------------------------------
extern "C" void kernel_launch(void* const* d_in, const int* in_sizes, int n_in,
                              void* d_out, int out_size, void* d_ws, size_t ws_size,
                              hipStream_t stream) {
    const float* x     = (const float*)d_in[0];   // (N, 64)
    // d_in[1] = edge_index: unused by the reference computation
    const int*   nidx  = (const int*)d_in[2];     // (G, M, 2, S)
    const float* W_att = (const float*)d_in[3];   // (64, 512)
    const float* att   = (const float*)d_in[4];   // (8, 128)
    float*       out   = (float*)d_out;           // (G, M, M, 8)

    // workspace layout: [0, 4KB) Bpk (f16, WMMA B layout), [4KB, ...) skv
    _Float16* Bpk = (_Float16*)d_ws;
    float*    skv = (float*)((char*)d_ws + 4096);

    // 1) fold weights into WMMA-B-layout: 1024 outputs
    fold_weights_kernel<<<4, 256, 0, stream>>>(W_att, att, Bpk);

    // 2) WMMA GEMM: 12500 tiles, 8 waves (tiles) per 256-thread block
    const int gemm_blocks = (TILES + WAVES_PB - 1) / WAVES_PB;   // 1563
    skv_gemm_kernel<<<gemm_blocks, 256, 0, stream>>>(x, Bpk, skv);

    // 3) gather / score / softmax: 131072 threads
    const int sc_threads = G_DIM * M_DIM * H_DIM;
    score_kernel<<<(sc_threads + 255) / 256, 256, 0, stream>>>(skv, nidx, out);
}
